// ExpressionClassifier_82703890252076
// MI455X (gfx1250) — matrix operand order repaired and checked by reading, not compiled, not run on hardware
//
#include <hip/hip_runtime.h>

#define T_FRAMES 2048
#define NN 68
#define EE 544
#define ET (EE + NN)      // 612 edges incl. self loops
#define HEADS 4
#define HID 32
#define F1 128            // HEADS*HID
#define GOUT 128
#define LH 128
#define NC 7
#define NEG 0.2f

typedef _Float16 v16h __attribute__((ext_vector_type(16)));
typedef _Float16 h8   __attribute__((ext_vector_type(8)));
typedef float    v8f  __attribute__((ext_vector_type(8)));

// LDS strides (in elements) chosen for bank-conflict-free WMMA fragment loads
#define H1S 136   // f16, row stride for h1 (A matrix source)
#define W2S 144   // f16, row stride for W2^T (B matrix source)
#define FTS 132   // f32, row stride for feature / hh buffer

// ---------------- workspace layout (bytes) ----------------
#define WS_CSR_OFF 0          //   69 int
#define WS_CSR_SRC 512        //  612 int
#define WS_W2T     4096       // 16384 f16 (W2 transposed, n-major)
#define WS_WIHT    40960      // 65536 f32 (Wih transposed, k-major)
#define WS_XG      303104     // 2048*512 f32 precomputed input gates

// =====================================================================
// Kernel 0: build CSR by destination, transpose/convert weights
// =====================================================================
__global__ __launch_bounds__(256) void prep_kernel(
    const int* __restrict__ ei, const float* __restrict__ W2,
    const float* __restrict__ Wih,
    int* __restrict__ csr_off, int* __restrict__ csr_src,
    _Float16* __restrict__ w2t, float* __restrict__ wihT) {
  __shared__ int s_cnt[NN];
  __shared__ int s_off[NN + 1];
  const int tid = threadIdx.x;

  if (tid < NN) s_cnt[tid] = 0;
  __syncthreads();
  for (int i = tid; i < ET; i += 256) {
    int dst = (i < EE) ? ei[EE + i] : (i - EE);
    atomicAdd(&s_cnt[dst], 1);
  }
  __syncthreads();
  if (tid == 0) {
    int acc = 0;
    for (int d = 0; d < NN; ++d) { s_off[d] = acc; acc += s_cnt[d]; }
    s_off[NN] = acc;
  }
  __syncthreads();
  if (tid < NN) s_cnt[tid] = s_off[tid];   // reuse as placement cursor
  __syncthreads();
  for (int i = tid; i < ET; i += 256) {
    int src = (i < EE) ? ei[i]      : (i - EE);
    int dst = (i < EE) ? ei[EE + i] : (i - EE);
    int pos = atomicAdd(&s_cnt[dst], 1);
    csr_src[pos] = src;
  }
  if (tid <= NN) csr_off[tid] = s_off[tid];

  // W2 (k-major [128,128]) -> f16 transposed (n-major, k contiguous)
  for (int i = tid; i < 128 * 128; i += 256) {
    int k = i >> 7, n = i & 127;
    w2t[n * 128 + k] = (_Float16)W2[k * 128 + n];
  }
  // Wih (gate-major [512,128]) -> transposed (k-major [128,512])
  for (int i = tid; i < 512 * 128; i += 256) {
    int g = i >> 7, k = i & 127;
    wihT[k * 512 + g] = Wih[g * 128 + k];
  }
}

// =====================================================================
// Kernel 1: per-frame 2-layer GAT encoder (1 block = 1 frame, 8 waves)
//           layer-2 transform done with v_wmma_f32_16x16x32_f16;
//           W2^T staged Global->LDS with async copy (ASYNCcnt) that
//           overlaps the whole layer-1 phase.
// =====================================================================
__global__ __launch_bounds__(256) void gat_kernel(
    const float* __restrict__ x,
    const float* __restrict__ W1, const float* __restrict__ a_src1,
    const float* __restrict__ a_dst1, const float* __restrict__ b1,
    const float* __restrict__ a_src2, const float* __restrict__ a_dst2,
    const float* __restrict__ b2,
    const int* __restrict__ csr_off, const int* __restrict__ csr_src,
    const _Float16* __restrict__ w2t,
    float* __restrict__ enc_out) {
  const int t = blockIdx.x;
  const int tid = threadIdx.x;

  __shared__ float s_x[NN][2];
  __shared__ float s_feat[80][FTS];                 // hp (layer1) then hh (layer2)
  __shared__ float s_as1[NN][HEADS], s_ad1[NN][HEADS];
  __shared__ alignas(32) _Float16 s_h1[80][H1S];    // layer1 output, f16, zero-padded rows
  __shared__ alignas(32) _Float16 s_w2t[128][W2S];  // B operand (n-major, k contiguous)
  __shared__ float s_as2[NN], s_ad2[NN];
  __shared__ int s_off[NN + 1];
  __shared__ int s_src[ET];
  __shared__ float s_wa1s[F1], s_wa1d[F1], s_b1[F1], s_wa2s[GOUT], s_wa2d[GOUT];
  __shared__ float s_pool[2][GOUT];

  // ---- kick off async Global->LDS copy of W2^T (32KB, 16B per op) ----
  // 2048 b128 chunks / 256 threads = 8 per thread.  GVS addressing:
  // addr = SGPR64(w2t) + VGPR32(byte offset);  VDST VGPR = LDS byte address.
  {
    const unsigned lds_base = (unsigned)(uintptr_t)&s_w2t[0][0];
    #pragma unroll
    for (int j = 0; j < 8; ++j) {
      int i = tid + j * 256;
      int n = i >> 4, c = i & 15;                       // row n, 16B chunk c
      unsigned ldsoff = lds_base + (unsigned)((n * W2S + c * 8) * 2);
      unsigned goff = (unsigned)((n * 128 + c * 8) * 2);
      asm volatile("global_load_async_to_lds_b128 %0, %1, %2"
                   :: "v"(ldsoff), "v"(goff), "s"(w2t) : "memory");
    }
  }

  // ---- stage small constants (overlaps the in-flight async copy) ----
  for (int i = tid; i < NN * 2; i += 256) ((float*)s_x)[i] = x[(size_t)t * NN * 2 + i];
  for (int i = tid; i <= NN; i += 256) s_off[i] = csr_off[i];
  for (int i = tid; i < ET; i += 256) s_src[i] = csr_src[i];
  for (int i = tid; i < F1; i += 256) {
    s_wa1s[i] = a_src1[i]; s_wa1d[i] = a_dst1[i]; s_b1[i] = b1[i];
    s_wa2s[i] = a_src2[i]; s_wa2d[i] = a_dst2[i];
  }
  __syncthreads();

  // ---- layer1 transform: hp = x @ W1  ([68,2]x[2,128]) ----
  for (int i = tid; i < NN * F1; i += 256) {
    int n = i >> 7, f = i & 127;
    s_feat[n][f] = s_x[n][0] * W1[f] + s_x[n][1] * W1[128 + f];
  }
  __syncthreads();

  // ---- per-(node,head) attention projections ----
  for (int i = tid; i < NN * HEADS; i += 256) {
    int n = i >> 2, h = i & 3;
    float as = 0.f, ad = 0.f;
    for (int c = 0; c < HID; ++c) {
      float v = s_feat[n][h * HID + c];
      as += v * s_wa1s[h * HID + c];
      ad += v * s_wa1d[h * HID + c];
    }
    s_as1[n][h] = as; s_ad1[n][h] = ad;
  }
  __syncthreads();

  // ---- layer1 softmax-aggregate, +b1, relu -> f16 ----
  {
    int f = tid & 127, half = tid >> 7, h = f >> 5;
    for (int dst = half; dst < NN; dst += 2) {
      int beg = s_off[dst], end = s_off[dst + 1];
      float adv = s_ad1[dst][h];
      float m = -3.4e38f;
      for (int j = beg; j < end; ++j) {
        float sc = s_as1[s_src[j]][h] + adv;
        sc = sc > 0.f ? sc : NEG * sc;
        m = fmaxf(m, sc);
      }
      float ssum = 0.f, acc = 0.f;
      for (int j = beg; j < end; ++j) {
        int sn = s_src[j];
        float sc = s_as1[sn][h] + adv;
        sc = sc > 0.f ? sc : NEG * sc;
        float ex = expf(sc - m);
        ssum += ex;
        acc += ex * s_feat[sn][f];
      }
      float o = acc / (ssum + 1e-16f) + s_b1[f];
      s_h1[dst][f] = (_Float16)(o > 0.f ? o : 0.f);
    }
  }
  // zero-pad rows 68..79 (read by WMMA A fragments)
  for (int i = tid; i < 12 * H1S; i += 256) s_h1[68 + i / H1S][i % H1S] = (_Float16)0.f;

  // async W2^T copy must be complete (per-lane) before the barrier below,
  // so after the barrier every thread sees the staged B operand.
  asm volatile("s_wait_asynccnt 0x0" ::: "memory");
  __syncthreads();

  // ---- layer2 transform: hh = h1 @ W2 via WMMA f32<=f16 ----
  {
    const int wave = tid >> 5, lane = tid & 31, lr = lane & 15, hi = lane >> 4;
    for (int tile = wave; tile < 40; tile += 8) {
      const int m0 = (tile >> 3) * 16, n0 = (tile & 7) * 16;
      v8f acc = {0.f, 0.f, 0.f, 0.f, 0.f, 0.f, 0.f, 0.f};
      #pragma unroll
      for (int kt = 0; kt < 4; ++kt) {
        const int k0 = kt * 32;
        // A 16x32 f16: lane lr (hi half selects K groups {0-7,16-23} / {8-15,24-31})
        h8 alo = *(const h8*)&s_h1[m0 + lr][k0 + hi * 8];
        h8 ahi = *(const h8*)&s_h1[m0 + lr][k0 + 16 + hi * 8];
        v16h a;
        #pragma unroll
        for (int e = 0; e < 8; ++e) { a[e] = alo[e]; a[8 + e] = ahi[e]; }
        // B 32x16 f16: lane holds N=n0+lr, K groups {0-7,16-23} / {8-15,24-31} (matches A)
        h8 blo = *(const h8*)&s_w2t[n0 + lr][k0 + hi * 8];
        h8 bhi = *(const h8*)&s_w2t[n0 + lr][k0 + 16 + hi * 8];
        v16h b;
        #pragma unroll
        for (int e = 0; e < 8; ++e) { b[e] = blo[e]; b[8 + e] = bhi[e]; }
        acc = __builtin_amdgcn_wmma_f32_16x16x32_f16(
            false, a, false, b, (short)0, acc, false, false);
      }
      #pragma unroll
      for (int v = 0; v < 8; ++v) s_feat[m0 + hi * 8 + v][n0 + lr] = acc[v];
    }
  }
  __syncthreads();

  // ---- layer2 attention projections ----
  for (int i = tid; i < NN * 2; i += 256) {
    int n = i >> 1;
    float sum = 0.f;
    if (i & 1) {
      for (int c = 0; c < GOUT; ++c) sum += s_feat[n][c] * s_wa2d[c];
      s_ad2[n] = sum;
    } else {
      for (int c = 0; c < GOUT; ++c) sum += s_feat[n][c] * s_wa2s[c];
      s_as2[n] = sum;
    }
  }
  __syncthreads();

  // ---- layer2 softmax-aggregate + global mean pool ----
  {
    int c = tid & 127, half = tid >> 7;
    float pool = 0.f;
    for (int dst = half; dst < NN; dst += 2) {
      int beg = s_off[dst], end = s_off[dst + 1];
      float adv = s_ad2[dst];
      float m = -3.4e38f;
      for (int j = beg; j < end; ++j) {
        float sc = s_as2[s_src[j]] + adv;
        sc = sc > 0.f ? sc : NEG * sc;
        m = fmaxf(m, sc);
      }
      float ssum = 0.f, acc = 0.f;
      for (int j = beg; j < end; ++j) {
        int sn = s_src[j];
        float sc = s_as2[sn] + adv;
        sc = sc > 0.f ? sc : NEG * sc;
        float ex = expf(sc - m);
        ssum += ex;
        acc += ex * s_feat[sn][c];
      }
      pool += acc / (ssum + 1e-16f);
    }
    s_pool[half][c] = pool;
  }
  __syncthreads();
  if (tid < GOUT) {
    float e = (s_pool[0][tid] + s_pool[1][tid]) * (1.f / NN) + b2[tid];
    enc_out[(size_t)t * GOUT + tid] = e;
  }
}

// =====================================================================
// Kernel 2: precompute input gates Xg[t] = WihT^T enc[t] + bih + bhh
// =====================================================================
__global__ __launch_bounds__(512) void xg_kernel(
    const float* __restrict__ enc, const float* __restrict__ wihT,
    const float* __restrict__ bih, const float* __restrict__ bhh,
    float* __restrict__ xg) {
  __shared__ float s_e[GOUT];
  const int t = blockIdx.x, tid = threadIdx.x;
  if (tid < GOUT) s_e[tid] = enc[(size_t)t * GOUT + tid];
  __syncthreads();
  float acc = bih[tid] + bhh[tid];
  for (int k = 0; k < GOUT; ++k) acc += wihT[k * 512 + tid] * s_e[k];
  xg[(size_t)t * 512 + tid] = acc;
}

// =====================================================================
// Kernel 3: sequential LSTM scan (WhhT resident in 256KB LDS) + FC head
// =====================================================================
__global__ __launch_bounds__(512) void lstm_kernel(
    const float* __restrict__ xg, const float* __restrict__ whh,
    const float* __restrict__ wfc, const float* __restrict__ bfc,
    float* __restrict__ out) {
  __shared__ float s_whhT[128 * 512];   // k-major: lane -> bank conflict-free
  __shared__ float s_h[LH], s_c[LH], s_g[512];
  const int tid = threadIdx.x;
  for (int i = tid; i < 128 * 512; i += 512) {
    int k = i >> 9, r = i & 511;
    s_whhT[i] = whh[r * 128 + k];
  }
  if (tid < LH) { s_h[tid] = 0.f; s_c[tid] = 0.f; }
  __syncthreads();
  for (int t = 0; t < T_FRAMES; ++t) {
    float acc = xg[(size_t)t * 512 + tid];
    #pragma unroll 8
    for (int k = 0; k < 128; ++k) acc += s_whhT[k * 512 + tid] * s_h[k];
    s_g[tid] = acc;
    __syncthreads();
    if (tid < LH) {
      float gi = s_g[tid], gf = s_g[128 + tid], gg = s_g[256 + tid], go = s_g[384 + tid];
      float si = 1.f / (1.f + expf(-gi));
      float sf = 1.f / (1.f + expf(-gf));
      float so = 1.f / (1.f + expf(-go));
      float c = sf * s_c[tid] + si * tanhf(gg);
      s_c[tid] = c;
      s_h[tid] = so * tanhf(c);
    }
    __syncthreads();
  }
  if (tid < NC) {
    float acc = bfc[tid];
    for (int k = 0; k < LH; ++k) acc += wfc[tid * 128 + k] * s_h[k];
    out[tid] = acc;
  }
}

// =====================================================================
extern "C" void kernel_launch(void* const* d_in, const int* in_sizes, int n_in,
                              void* d_out, int out_size, void* d_ws, size_t ws_size,
                              hipStream_t stream) {
  (void)in_sizes; (void)n_in; (void)out_size; (void)ws_size;
  const float* x    = (const float*)d_in[0];
  const int*   ei   = (const int*)d_in[1];
  const float* W1   = (const float*)d_in[2];
  const float* a_s1 = (const float*)d_in[3];
  const float* a_d1 = (const float*)d_in[4];
  const float* b1   = (const float*)d_in[5];
  const float* W2   = (const float*)d_in[6];
  const float* a_s2 = (const float*)d_in[7];
  const float* a_d2 = (const float*)d_in[8];
  const float* b2   = (const float*)d_in[9];
  const float* Wih  = (const float*)d_in[10];
  const float* Whh  = (const float*)d_in[11];
  const float* bih  = (const float*)d_in[12];
  const float* bhh  = (const float*)d_in[13];
  const float* Wfc  = (const float*)d_in[14];
  const float* bfc  = (const float*)d_in[15];

  char* ws = (char*)d_ws;
  int*      csr_off = (int*)(ws + WS_CSR_OFF);
  int*      csr_src = (int*)(ws + WS_CSR_SRC);
  _Float16* w2t     = (_Float16*)(ws + WS_W2T);
  float*    wihT    = (float*)(ws + WS_WIHT);
  float*    xg      = (float*)(ws + WS_XG);

  float* out = (float*)d_out;
  float* enc = out + NC;   // output order: logits[7], enc[2048*128]

  prep_kernel<<<1, 256, 0, stream>>>(ei, W2, Wih, csr_off, csr_src, w2t, wihT);
  gat_kernel<<<T_FRAMES, 256, 0, stream>>>(x, W1, a_s1, a_d1, b1, a_s2, a_d2, b2,
                                           csr_off, csr_src, w2t, enc);
  xg_kernel<<<T_FRAMES, 512, 0, stream>>>(enc, wihT, bih, bhh, xg);
  lstm_kernel<<<1, 512, 0, stream>>>(xg, Whh, Wfc, bfc, out);
}
